// LlamaAttention_44753559225111
// MI455X (gfx1250) — matrix operand order repaired and checked by reading, not compiled, not run on hardware
//
#include <hip/hip_runtime.h>

// ---------------------------------------------------------------------------
// Llama attention block (QKV proj + RoPE + causal GQA attention + O proj)
// for MI455X / gfx1250.  All three GEMM stages run on v_wmma_f32_16x16x32_bf16
// (f32 accumulate).  fp32 inputs are converted to bf16 once; weight panels are
// stored transposed ([N,K]) so B-fragments are two 8-elem K-runs (k, k+16).
// Global streams are prefetched ahead (global_prefetch_b8) into L2/WGP$.
// Workspace layout (byte offsets), total 176MB assumed available in d_ws:
//   [0,32MB)    hidden_bf16            -> reused as attn_out after GEMM1
//   [32,80MB)   w_qkv^T bf16 [6144,4096] -> reused as w_o^T [4096,4096]
//   [80,128MB)  qkv bf16 [4096,6144]
//   [128,160MB) Q bf16 [B*H, S, D]   (1/sqrt(D) folded in)
//   [160,168MB) K bf16 [B*Hkv, S, D]
//   [168,176MB) V^T bf16 [B*Hkv, D, S]
// ---------------------------------------------------------------------------

typedef __bf16 bf16_t;
typedef __attribute__((ext_vector_type(16))) __bf16 v16bf;
typedef __attribute__((ext_vector_type(8)))  __bf16 v8bf;
typedef __attribute__((ext_vector_type(4)))  __bf16 v4bf;
typedef __attribute__((ext_vector_type(8)))  float  v8f;
typedef __attribute__((ext_vector_type(4)))  float  v4f;

union ABfrag { v16bf v; v8bf h[2]; };

__device__ __forceinline__ v8f wmma_bf16(const ABfrag& a, const ABfrag& b, v8f c) {
  return __builtin_amdgcn_wmma_f32_16x16x32_bf16(false, a.v, false, b.v,
                                                 (short)0, c, false, false);
}

#define LOG2E 1.4426950408889634f
#define NEG_BIG (-1.0e30f)

// --------------------------- f32 -> bf16 convert ---------------------------
__global__ void __launch_bounds__(256) cvt_f32_bf16(const float* __restrict__ in,
                                                    bf16_t* __restrict__ out, int n) {
  int i = (blockIdx.x * 256 + threadIdx.x) * 4;
  if (i < n) {
    v4f x = *(const v4f*)(in + i);
    v4bf y;
    #pragma unroll
    for (int j = 0; j < 4; ++j) y[j] = (bf16_t)x[j];
    *(v4bf*)(out + i) = y;
  }
}

// ---------------- f32 [R,C] -> bf16 [C,R] transpose-convert ----------------
__global__ void __launch_bounds__(256) transpose_cvt(const float* __restrict__ in,
                                                     bf16_t* __restrict__ out,
                                                     int R, int C) {
  __shared__ float tile[32][33];
  int c0 = blockIdx.x * 32, r0 = blockIdx.y * 32;
  int tx = threadIdx.x, ty = threadIdx.y;  // (32,8)
  #pragma unroll
  for (int j = 0; j < 4; ++j)
    tile[ty + j * 8][tx] = in[(size_t)(r0 + ty + j * 8) * C + c0 + tx];
  __syncthreads();
  #pragma unroll
  for (int j = 0; j < 4; ++j)
    out[(size_t)(c0 + ty + j * 8) * R + r0 + tx] = (bf16_t)tile[tx][ty + j * 8];
}

// --------------------- WMMA GEMM: C = A[M,K] * B^T[N,K] --------------------
// 128x128 block tile, BK=64, 8 waves of 32(M)x64(N), double-buffered LDS.
template <bool OUT_F32>
__global__ void __launch_bounds__(256) gemm_bt_kernel(const bf16_t* __restrict__ A,
                                                      const bf16_t* __restrict__ BT,
                                                      void* __restrict__ Cout,
                                                      int M, int N, int K) {
  constexpr int BM = 128, BN = 128, BK = 64, LDT = BK + 8;  // 72 halfwords/row
  __shared__ bf16_t As[2][BM * LDT];
  __shared__ bf16_t Bs[2][BN * LDT];

  const int tid  = threadIdx.x;
  const int wave = tid >> 5, lane = tid & 31;
  const int half = lane >> 4, l15 = lane & 15;
  const int wm = wave & 3, wn = wave >> 2;
  const int m0 = blockIdx.x * BM, n0 = blockIdx.y * BN;

  const int rowL = tid >> 1;
  const int segL = (tid & 1) * 32;
  const bf16_t* gA = A  + (size_t)(m0 + rowL) * K + segL;
  const bf16_t* gB = BT + (size_t)(n0 + rowL) * K + segL;

  v8f acc[2][4] = {};
  v8bf la[4], lb[4];

  auto load_g = [&](int kt) {
    const bf16_t* pa = gA + (size_t)kt * BK;
    const bf16_t* pb = gB + (size_t)kt * BK;
    #pragma unroll
    for (int i = 0; i < 4; ++i) {
      la[i] = *(const v8bf*)(pa + i * 8);
      lb[i] = *(const v8bf*)(pb + i * 8);
    }
  };
  auto store_s = [&](int buf) {
    bf16_t* sa = &As[buf][rowL * LDT + segL];
    bf16_t* sb = &Bs[buf][rowL * LDT + segL];
    #pragma unroll
    for (int i = 0; i < 4; ++i) {
      *(v8bf*)(sa + i * 8) = la[i];
      *(v8bf*)(sb + i * 8) = lb[i];
    }
  };
  auto compute = [&](int buf) {
    const bf16_t* as = As[buf];
    const bf16_t* bs = Bs[buf];
    #pragma unroll
    for (int ks = 0; ks < 2; ++ks) {
      const int k0 = ks * 32;
      ABfrag af[2], bfr[4];
      #pragma unroll
      for (int mf = 0; mf < 2; ++mf) {
        const int row = wm * 32 + mf * 16 + l15;
        af[mf].h[0] = *(const v8bf*)&as[row * LDT + k0 + half * 8];
        af[mf].h[1] = *(const v8bf*)&as[row * LDT + k0 + half * 8 + 16];
      }
      #pragma unroll
      for (int nf = 0; nf < 4; ++nf) {
        const int n = wn * 64 + nf * 16 + l15;
        bfr[nf].h[0] = *(const v8bf*)&bs[n * LDT + k0 + half * 8];
        bfr[nf].h[1] = *(const v8bf*)&bs[n * LDT + k0 + half * 8 + 16];
      }
      #pragma unroll
      for (int mf = 0; mf < 2; ++mf)
        #pragma unroll
        for (int nf = 0; nf < 4; ++nf)
          acc[mf][nf] = wmma_bf16(af[mf], bfr[nf], acc[mf][nf]);
    }
  };

  const int nkt = K / BK;
  load_g(0);
  store_s(0);
  __syncthreads();
  for (int kt = 0; kt < nkt; ++kt) {
    const int buf = kt & 1;
    const bool more = (kt + 1) < nkt;
    if (more) load_g(kt + 1);
    if (kt + 2 < nkt) {  // march the streams into L2/WGP$ two stages ahead
      __builtin_prefetch(gA + (size_t)(kt + 2) * BK, 0, 0);
      __builtin_prefetch(gB + (size_t)(kt + 2) * BK, 0, 0);
    }
    compute(buf);
    if (more) store_s(buf ^ 1);
    __syncthreads();
  }

  const int row0 = m0 + wm * 32, col0 = n0 + wn * 64;
  #pragma unroll
  for (int mf = 0; mf < 2; ++mf)
    #pragma unroll
    for (int nf = 0; nf < 4; ++nf)
      #pragma unroll
      for (int vr = 0; vr < 8; ++vr) {
        const int r = row0 + mf * 16 + vr + 8 * half;
        const int c = col0 + nf * 16 + l15;
        if (OUT_F32)
          ((float*)Cout)[(size_t)r * N + c] = acc[mf][nf][vr];
        else
          ((bf16_t*)Cout)[(size_t)r * N + c] = (bf16_t)acc[mf][nf][vr];
      }
}

// ------------------------- RoPE + QKV repack -------------------------------
// grid(T, 48), block(128).  Heads 0..31: Q (rope, *1/sqrt(D)); 32..39: K (rope);
// 40..47: V copied transposed to [.,D,S].
__global__ void __launch_bounds__(128) rope_repack(const bf16_t* __restrict__ qkv,
                                                   bf16_t* __restrict__ Q,
                                                   bf16_t* __restrict__ Kd,
                                                   bf16_t* __restrict__ VT) {
  constexpr int S = 1024, D = 128, H = 32, HKV = 8;
  const int t = blockIdx.x, head = blockIdx.y, d = threadIdx.x;
  const int b = t >> 10, s = t & 1023;
  const bf16_t* src = qkv + (size_t)t * 6144 + head * D;
  if (head < H + HKV) {
    const float x = (float)src[d];
    const float partner = (float)src[d ^ 64];
    const int d2 = d & 63;
    // inv_freq = 10000^(-d2/64) = exp2(-d2*log2(10000)/64)
    const float inv = exp2f(-(float)d2 * (13.287712379549449f / 64.0f));
    const float ang = (float)s * inv;
    const float c = __cosf(ang), sn = __sinf(ang);
    const float rot = (d < 64) ? -partner : partner;
    float y = x * c + rot * sn;
    if (head < H) {
      y *= 0.08838834764831845f;  // fold 1/sqrt(128) into Q
      Q[((size_t)(b * H + head) * S + s) * D + d] = (bf16_t)y;
    } else {
      Kd[((size_t)(b * HKV + (head - H)) * S + s) * D + d] = (bf16_t)y;
    }
  } else {
    const int hv = head - (H + HKV);
    VT[((size_t)(b * HKV + hv) * D + d) * S + s] = src[d];
  }
}

// ------------------- flash-style causal GQA attention ----------------------
// grid(8, 128): x = q-tile of 128 rows, y = b*32+h.  256 threads = 8 waves,
// each wave owns 16 q-rows x full D=128.  64-key tiles, online softmax.
__global__ void __launch_bounds__(256) attn_kernel(const bf16_t* __restrict__ Q,
                                                   const bf16_t* __restrict__ Kb,
                                                   const bf16_t* __restrict__ VT,
                                                   bf16_t* __restrict__ Oout) {
  constexpr int S = 1024, D = 128, HKV = 8;
  constexpr int LK = D + 8;   // 136 halfwords per K-tile row
  constexpr int LV = 64 + 8;  // 72  halfwords per V^T-tile row
  constexpr int LP = 64 + 8;  // 72  halfwords per P-tile row
  __shared__ bf16_t Ks[64 * LK];
  __shared__ bf16_t Vs[128 * LV];
  __shared__ bf16_t Ps[8 * 16 * LP];

  const int qt = blockIdx.x, bh = blockIdx.y;
  const int b = bh >> 5, h = bh & 31;
  const int kvh = b * HKV + (h >> 2);
  const int tid = threadIdx.x;
  const int wave = tid >> 5, lane = tid & 31;
  const int half = lane >> 4, l15 = lane & 15;
  const int qbase = qt * 128;

  // Q fragments for this wave's 16 rows (A-layout, direct from global)
  const bf16_t* Qg = Q + ((size_t)bh * S + qbase + wave * 16 + l15) * D;
  ABfrag qf[4];
  #pragma unroll
  for (int kf = 0; kf < 4; ++kf) {
    qf[kf].h[0] = *(const v8bf*)(Qg + kf * 32 + half * 8);
    qf[kf].h[1] = *(const v8bf*)(Qg + kf * 32 + half * 8 + 16);
  }

  v8f o[8] = {};
  float mrow[8], lrow[8];
  #pragma unroll
  for (int r = 0; r < 8; ++r) { mrow[r] = NEG_BIG; lrow[r] = 0.0f; }

  const bf16_t* Kg = Kb + (size_t)kvh * S * D;
  const bf16_t* Vg = VT + (size_t)kvh * D * S;
  const int nkt = (qt + 1) * 2;

  for (int kt = 0; kt < nkt; ++kt) {
    {  // cooperative K tile load: 64 keys x 128 D
      const int row = tid >> 2, seg = (tid & 3) * 32;
      const bf16_t* src = Kg + (size_t)(kt * 64 + row) * D + seg;
      bf16_t* dst = &Ks[row * LK + seg];
      #pragma unroll
      for (int i = 0; i < 4; ++i) *(v8bf*)(dst + i * 8) = *(const v8bf*)(src + i * 8);
      if (kt + 1 < nkt) __builtin_prefetch(src + (size_t)64 * D, 0, 0);
    }
    {  // cooperative V^T tile load: 128 D x 64 keys
      const int row = tid >> 1, seg = (tid & 1) * 32;
      const bf16_t* src = Vg + (size_t)row * S + kt * 64 + seg;
      bf16_t* dst = &Vs[row * LV + seg];
      #pragma unroll
      for (int i = 0; i < 4; ++i) *(v8bf*)(dst + i * 8) = *(const v8bf*)(src + i * 8);
      if (kt + 1 < nkt) __builtin_prefetch(src + 64, 0, 0);
    }
    __syncthreads();

    // scores: 4 column fragments of 16 keys, K-dim = D = 4 WMMA steps
    v8f sf[4];
    #pragma unroll
    for (int cf = 0; cf < 4; ++cf) {
      v8f c = {};
      const int key = cf * 16 + l15;
      #pragma unroll
      for (int kf = 0; kf < 4; ++kf) {
        ABfrag kfr;
        kfr.h[0] = *(const v8bf*)&Ks[key * LK + kf * 32 + half * 8];
        kfr.h[1] = *(const v8bf*)&Ks[key * LK + kf * 32 + half * 8 + 16];
        c = wmma_bf16(qf[kf], kfr, c);
      }
      sf[cf] = c;
    }

    // causal mask + online softmax update (rows live per lane-half)
    #pragma unroll
    for (int vr = 0; vr < 8; ++vr) {
      const int qrow = qbase + wave * 16 + vr + 8 * half;
      float mx = mrow[vr];
      #pragma unroll
      for (int cf = 0; cf < 4; ++cf) {
        const int key = kt * 64 + cf * 16 + l15;
        float sv = (key <= qrow) ? sf[cf][vr] : NEG_BIG;
        sf[cf][vr] = sv;
        mx = fmaxf(mx, sv);
      }
      #pragma unroll
      for (int msk = 1; msk <= 8; msk <<= 1) mx = fmaxf(mx, __shfl_xor(mx, msk, 32));

      const float alpha = exp2f((mrow[vr] - mx) * LOG2E);
      mrow[vr] = mx;
      lrow[vr] *= alpha;
      #pragma unroll
      for (int nf = 0; nf < 8; ++nf) o[nf][vr] *= alpha;

      float rs = 0.0f;
      #pragma unroll
      for (int cf = 0; cf < 4; ++cf) {
        const float p = exp2f((sf[cf][vr] - mx) * LOG2E);
        sf[cf][vr] = p;
        rs += p;
      }
      #pragma unroll
      for (int msk = 1; msk <= 8; msk <<= 1) rs += __shfl_xor(rs, msk, 32);
      lrow[vr] += rs;
    }

    // C-layout P -> bf16 A-layout via per-wave LDS staging tile
    bf16_t* pw = Ps + wave * 16 * LP;
    #pragma unroll
    for (int cf = 0; cf < 4; ++cf)
      #pragma unroll
      for (int vr = 0; vr < 8; ++vr)
        pw[(vr + 8 * half) * LP + cf * 16 + l15] = (bf16_t)sf[cf][vr];

    // O += P[16x64] * V[64x128]
    #pragma unroll
    for (int ks = 0; ks < 2; ++ks) {
      ABfrag pf;
      pf.h[0] = *(const v8bf*)&pw[l15 * LP + ks * 32 + half * 8];
      pf.h[1] = *(const v8bf*)&pw[l15 * LP + ks * 32 + half * 8 + 16];
      #pragma unroll
      for (int nf = 0; nf < 8; ++nf) {
        ABfrag vf;
        vf.h[0] = *(const v8bf*)&Vs[(nf * 16 + l15) * LV + ks * 32 + half * 8];
        vf.h[1] = *(const v8bf*)&Vs[(nf * 16 + l15) * LV + ks * 32 + half * 8 + 16];
        o[nf] = wmma_bf16(pf, vf, o[nf]);
      }
    }
    __syncthreads();
  }

  // normalize and write [T, H*D] bf16
  #pragma unroll
  for (int vr = 0; vr < 8; ++vr) {
    const float inv = 1.0f / lrow[vr];
    const int t = b * S + qbase + wave * 16 + vr + 8 * half;
    #pragma unroll
    for (int nf = 0; nf < 8; ++nf)
      Oout[(size_t)t * 4096 + h * 128 + nf * 16 + l15] = (bf16_t)(o[nf][vr] * inv);
  }
}

// ---------------------------------------------------------------------------
extern "C" void kernel_launch(void* const* d_in, const int* in_sizes, int n_in,
                              void* d_out, int out_size, void* d_ws, size_t ws_size,
                              hipStream_t stream) {
  (void)in_sizes; (void)n_in; (void)out_size; (void)ws_size;
  const float* hidden = (const float*)d_in[0];  // [4096, 4096]
  const float* w_qkv  = (const float*)d_in[1];  // [4096, 6144]
  const float* w_o    = (const float*)d_in[2];  // [4096, 4096]
  float* out = (float*)d_out;                   // [4096, 4096]

  char* ws = (char*)d_ws;
  bf16_t* hid_bf = (bf16_t*)(ws);                           // 32MB (reused: attn_out)
  bf16_t* w_bt   = (bf16_t*)(ws + (size_t)32  * (1 << 20)); // 48MB (w_qkv^T / w_o^T)
  bf16_t* qkv    = (bf16_t*)(ws + (size_t)80  * (1 << 20)); // 48MB
  bf16_t* Qb     = (bf16_t*)(ws + (size_t)128 * (1 << 20)); // 32MB
  bf16_t* Kb     = (bf16_t*)(ws + (size_t)160 * (1 << 20)); // 8MB
  bf16_t* VTb    = (bf16_t*)(ws + (size_t)168 * (1 << 20)); // 8MB

  // 1. convert inputs to bf16 (weights transposed to [N,K])
  cvt_f32_bf16<<<16384, 256, 0, stream>>>(hidden, hid_bf, 4096 * 4096);
  transpose_cvt<<<dim3(192, 128), dim3(32, 8), 0, stream>>>(w_qkv, w_bt, 4096, 6144);

  // 2. QKV projection: [4096,4096] x [4096,6144] -> bf16
  gemm_bt_kernel<false><<<dim3(32, 48), 256, 0, stream>>>(hid_bf, w_bt, qkv,
                                                          4096, 6144, 4096);

  // 3. RoPE + repack into Q / K / V^T
  rope_repack<<<dim3(4096, 48), 128, 0, stream>>>(qkv, Qb, Kb, VTb);

  // 4. w_o^T (reuses the w_qkv^T region; safe: stream-ordered after GEMM1)
  transpose_cvt<<<dim3(128, 128), dim3(32, 8), 0, stream>>>(w_o, w_bt, 4096, 4096);

  // 5. causal GQA attention -> attn_out bf16 (reuses hidden_bf16 region)
  bf16_t* attn = hid_bf;
  attn_kernel<<<dim3(8, 128), 256, 0, stream>>>(Qb, Kb, VTb, attn);

  // 6. output projection -> f32 d_out
  gemm_bt_kernel<true><<<dim3(32, 32), 256, 0, stream>>>(attn, w_bt, out,
                                                         4096, 4096, 4096);
}
